// MoEGate_65060164600321
// MI455X (gfx1250) — matrix operand order repaired and checked by reading, not compiled, not run on hardware
//
#include <hip/hip_runtime.h>
#include <hip/hip_bf16.h>

typedef __attribute__((ext_vector_type(16))) _Float16 v16h;
typedef __attribute__((ext_vector_type(8)))  _Float16 v8h;
typedef __attribute__((ext_vector_type(8)))  float    v8f;

#define HDIM   256
#define NEXP   32
#define SEQLEN 8192
#define NTOK   (16 * 8192)      // 131072 tokens
#define TPB    128              // tokens per block
#define NWAVE  8                // 256 threads / wave32
#define XSTR   (HDIM + 8)       // padded f16 row stride (264) -> conflict-free frag reads
#define LSTR   (NEXP + 1)       // padded logit row stride (33 floats)

// ---------------------------------------------------------------------------
// Zero the per-(batch, expert) accumulators in workspace.
// ws layout: float score_sum[16*32] | int count[16*32]
// ---------------------------------------------------------------------------
__global__ void init_ws_kernel(float* __restrict__ ws_score, int* __restrict__ ws_count) {
    int t = threadIdx.x;   // 512 threads
    ws_score[t] = 0.0f;
    ws_count[t] = 0;
}

// ---------------------------------------------------------------------------
// Main gate kernel: logits via WMMA f16 (f32 accumulate), softmax, top-1,
// per-block accumulation of aux-loss statistics.
// ---------------------------------------------------------------------------
__global__ __launch_bounds__(256) void moe_gate_kernel(
    const float* __restrict__ x,        // [NTOK, 256] f32
    const float* __restrict__ w,        // [32, 256]  f32
    float* __restrict__ out,            // [NTOK idx | NTOK weight | 1 aux]
    float* __restrict__ ws_score,       // [16*32]
    int*   __restrict__ ws_count)       // [16*32]
{
    __shared__ __align__(16) _Float16 ldsX[TPB * XSTR];     // 67.5 KB
    __shared__ __align__(16) _Float16 ldsW[NEXP * XSTR];    // 16.9 KB
    __shared__ __align__(16) float    ldsL[NWAVE * 16 * LSTR]; // 16.9 KB
    __shared__ float scoreAcc[NEXP];
    __shared__ int   countAcc[NEXP];

    const int tid      = threadIdx.x;
    const int lane     = tid & 31;
    const int wave     = tid >> 5;
    const int blockTok = blockIdx.x * TPB;

    if (tid < NEXP) { scoreAcc[tid] = 0.0f; countAcc[tid] = 0; }

    // ---- Stage x tile (128 rows x 256 f32) into LDS as f16, coalesced -----
    {
        const float4* x4 = (const float4*)(x + (size_t)blockTok * HDIM);
        #pragma unroll
        for (int i = 0; i < 32; ++i) {
            int idx = i * 256 + tid;          // 0..8191 float4s
            int row = idx >> 6;               // /64 float4 per row
            int c4  = idx & 63;
            float4 v = x4[idx];
            union { _Float16 h[4]; unsigned long long u; } p;
            p.h[0] = (_Float16)v.x; p.h[1] = (_Float16)v.y;
            p.h[2] = (_Float16)v.z; p.h[3] = (_Float16)v.w;
            *(unsigned long long*)&ldsX[row * XSTR + c4 * 4] = p.u;
        }
        // weight 32 x 256 f32 = 2048 float4
        const float4* w4 = (const float4*)w;
        #pragma unroll
        for (int i = 0; i < 8; ++i) {
            int idx = i * 256 + tid;
            int row = idx >> 6;
            int c4  = idx & 63;
            float4 v = w4[idx];
            union { _Float16 h[4]; unsigned long long u; } p;
            p.h[0] = (_Float16)v.x; p.h[1] = (_Float16)v.y;
            p.h[2] = (_Float16)v.z; p.h[3] = (_Float16)v.w;
            *(unsigned long long*)&ldsW[row * XSTR + c4 * 4] = p.u;
        }
    }
    __syncthreads();

    // ---- WMMA: this wave computes logits for tokens [wave*16, wave*16+16) --
    const int m    = lane & 15;     // A row / B column / D column
    const int half = lane >> 4;

    v8f acc0 = {};   // experts 0..15
    v8f acc1 = {};   // experts 16..31
    const _Float16* xrow  = &ldsX[(wave * 16 + m) * XSTR];
    const _Float16* wrow0 = &ldsW[m * XSTR];
    const _Float16* wrow1 = &ldsW[(16 + m) * XSTR];

    #pragma unroll
    for (int kk = 0; kk < 8; ++kk) {
        const int k0 = kk * 32;
        // A 16x32 f16, contiguous k order used here: lanes 0-15 elems0-15=K0-15;
        //                     lanes16-31 elems0-15=K16-31
        v8h alo = *(const v8h*)(xrow + k0 + half * 16);
        v8h ahi = *(const v8h*)(xrow + k0 + half * 16 + 8);
        v16h a = __builtin_shufflevector(alo, ahi,
                  0,1,2,3,4,5,6,7,8,9,10,11,12,13,14,15);
        // B 32x16 f16, same contiguous k order as A: lanes 0-15 K0-15, lanes 16-31 K16-31; N=lane%16
        const int kb = k0 + half * 16;
        v8h b0lo = *(const v8h*)(wrow0 + kb);
        v8h b0hi = *(const v8h*)(wrow0 + kb + 8);
        v16h b0 = __builtin_shufflevector(b0lo, b0hi,
                  0,1,2,3,4,5,6,7,8,9,10,11,12,13,14,15);
        v8h b1lo = *(const v8h*)(wrow1 + kb);
        v8h b1hi = *(const v8h*)(wrow1 + kb + 8);
        v16h b1 = __builtin_shufflevector(b1lo, b1hi,
                  0,1,2,3,4,5,6,7,8,9,10,11,12,13,14,15);

        acc0 = __builtin_amdgcn_wmma_f32_16x16x32_f16(
                   false, a, false, b0, (short)0, acc0, false, false);
        acc1 = __builtin_amdgcn_wmma_f32_16x16x32_f16(
                   false, a, false, b1, (short)0, acc1, false, false);
    }

    // ---- Spill logits to LDS: D VGPR r -> token row r + 8*half, col = m ----
    {
        float* lrow = &ldsL[wave * 16 * LSTR];
        #pragma unroll
        for (int r = 0; r < 8; ++r) {
            int mr = r + half * 8;
            lrow[mr * LSTR + m]      = acc0[r];
            lrow[mr * LSTR + 16 + m] = acc1[r];
        }
    }
    __syncthreads();

    // ---- Softmax + top-1 per token (lanes 0-15 of each wave) --------------
    if (lane < 16) {
        float* lr = &ldsL[(wave * 16 + lane) * LSTR];
        float maxv = -3.402823466e+38f;
        int   arg  = 0;
        #pragma unroll
        for (int e = 0; e < NEXP; ++e) {
            float v = lr[e];
            if (v > maxv) { maxv = v; arg = e; }    // first max, like jax top_k
        }
        float sum = 0.0f;
        #pragma unroll
        for (int e = 0; e < NEXP; ++e) sum += __expf(lr[e] - maxv);
        float inv = 1.0f / sum;

        int tok = blockTok + wave * 16 + lane;
        out[tok]        = (float)arg;       // topk_idx
        out[NTOK + tok] = inv;              // topk_weight = exp(0)/sum, SCALING=1
        #pragma unroll
        for (int e = 0; e < NEXP; ++e) lr[e] = __expf(lr[e] - maxv) * inv;
        atomicAdd(&countAcc[arg], 1);
    }
    __syncthreads();

    // ---- Per-block score sums: thread group g sums its wave's 16 tokens ---
    {
        int e = tid & 31;
        int g = tid >> 5;
        const float* lg = &ldsL[g * 16 * LSTR];
        float s = 0.0f;
        #pragma unroll
        for (int mm = 0; mm < 16; ++mm) s += lg[mm * LSTR + e];
        atomicAdd(&scoreAcc[e], s);
    }
    __syncthreads();

    // ---- Flush block accumulators to global (one batch per block) ---------
    if (tid < NEXP) {
        int b = blockIdx.x >> 6;    // 64 blocks of 128 tokens per 8192-seq batch
        atomicAdd(&ws_score[b * NEXP + tid], scoreAcc[tid]);
        atomicAdd(&ws_count[b * NEXP + tid], countAcc[tid]);
    }
}

// ---------------------------------------------------------------------------
// Aux loss reduction over 16 batches x 32 experts.
// ---------------------------------------------------------------------------
__global__ void aux_loss_kernel(const float* __restrict__ ws_score,
                                const int* __restrict__ ws_count,
                                float* __restrict__ out_aux) {
    __shared__ float red[512];
    int t = threadIdx.x;     // 512 = 16*32
    float ce   = (float)ws_count[t] * (32.0f / 8192.0f);
    float mean = ws_score[t] * (1.0f / 8192.0f);
    red[t] = ce * mean;
    __syncthreads();
    for (int s = 256; s > 0; s >>= 1) {
        if (t < s) red[t] += red[t + s];
        __syncthreads();
    }
    if (t == 0) out_aux[0] = 0.001f * red[0] / 16.0f;
}

// ---------------------------------------------------------------------------
extern "C" void kernel_launch(void* const* d_in, const int* in_sizes, int n_in,
                              void* d_out, int out_size, void* d_ws, size_t ws_size,
                              hipStream_t stream) {
    const float* x = (const float*)d_in[0];   // hidden_states [16,8192,256] f32
    const float* w = (const float*)d_in[1];   // weight [32,256] f32
    float* out = (float*)d_out;               // [N idx | N weight | aux]

    float* ws_score = (float*)d_ws;
    int*   ws_count = (int*)d_ws + 512;

    init_ws_kernel<<<1, 512, 0, stream>>>(ws_score, ws_count);
    moe_gate_kernel<<<NTOK / TPB, 256, 0, stream>>>(x, w, out, ws_score, ws_count);
    aux_loss_kernel<<<1, 512, 0, stream>>>(ws_score, ws_count, out + 2 * NTOK);
}
